// SpatiotemporalGNN_31439160607197
// MI455X (gfx1250) — matrix operand order repaired and checked by reading, not compiled, not run on hardware
//
#include <hip/hip_runtime.h>

#define F_IN  64
#define HID   128
#define NOUT  64
#define NB    8
#define NT    64
#define NN    512
#define NROWS (NB * NT * NN)   // 262144 tokens
#define BNROWS (NB * NN)       // 4096 GRU rows
#define AXPAD 40               // padded k-stride (halfs) for transposed staging

typedef __attribute__((ext_vector_type(16))) _Float16 v16h;
typedef __attribute__((ext_vector_type(8)))  _Float16 v8h;
typedef __attribute__((ext_vector_type(8)))  float    v8f;

// ---------------------------------------------------------------------------
// WMMA helpers (CDNA5 wave32, 16x16x32 f16 -> f32)
// A order used by this code (16-bit A 16x32): lane L: row M = L&15, hi = L>>4;
//   elem j -> K = hi*16 + j
// B order used by this code (32x16): lane L: col N = L&15, hi = L>>4; elem j -> K = hi*16 + j
// C/D layout (f32 16x16): VGPR r: lanes0-15 M=r, lanes16-31 M=r+8; N = L&15
// ---------------------------------------------------------------------------
__device__ __forceinline__ v8f wmma16(v16h a, v16h b, v8f c) {
  return __builtin_amdgcn_wmma_f32_16x16x32_f16(false, a, false, b, (short)0, c,
                                                false, false);
}

__device__ __forceinline__ v16h concat8(v8h lo, v8h hb) {
  return __builtin_shufflevector(lo, hb, 0, 1, 2, 3, 4, 5, 6, 7, 8, 9, 10, 11,
                                 12, 13, 14, 15);
}

__device__ __forceinline__ v16h load_a_tile(const _Float16* rowk, int hi) {
  v8h lo = *(const v8h*)(rowk + hi * 16);
  v8h hb = *(const v8h*)(rowk + hi * 16 + 8);
  return concat8(lo, hb);
}

__device__ __forceinline__ v8f splat8(float x) {
  v8f v;
#pragma unroll
  for (int r = 0; r < 8; ++r) v[r] = x;
  return v;
}

__device__ __forceinline__ float sigf(float x) { return 1.0f / (1.0f + __expf(-x)); }

// ---------------------------------------------------------------------------
// f32 -> f16 conversion (weights, x_seq)
// ---------------------------------------------------------------------------
__global__ void k_f32_to_f16(const float* __restrict__ src,
                             _Float16* __restrict__ dst, long n) {
  long i = (long)blockIdx.x * blockDim.x + threadIdx.x;
  long stride = (long)gridDim.x * blockDim.x;
  for (; i < n; i += stride) dst[i] = (_Float16)src[i];
}

// ---------------------------------------------------------------------------
// degree^-1/2 of A_hat = A + I
// ---------------------------------------------------------------------------
__global__ void k_deg(const float* __restrict__ A, float* __restrict__ dis) {
  __shared__ float red[256];
  const int row = blockIdx.x;
  float s = 0.f;
  for (int m = threadIdx.x; m < NN; m += 256) s += A[(long)row * NN + m];
  red[threadIdx.x] = s;
  __syncthreads();
  for (int w = 128; w > 0; w >>= 1) {
    if (threadIdx.x < w) red[threadIdx.x] += red[threadIdx.x + w];
    __syncthreads();
  }
  if (threadIdx.x == 0) dis[row] = rsqrtf(red[0] + 1.0f + 1e-12f);
}

// A_tilde = D^-1/2 (A+I) D^-1/2 in f16
__global__ void k_At16(const float* __restrict__ A, const float* __restrict__ dis,
                       _Float16* __restrict__ At) {
  const int row = blockIdx.x;
  const float dr = dis[row];
  for (int m = threadIdx.x; m < NN; m += 256) {
    float v = A[(long)row * NN + m] + ((m == row) ? 1.0f : 0.0f);
    At[(long)row * NN + m] = (_Float16)(dr * v * dis[m]);
  }
}

// ---------------------------------------------------------------------------
// Y(R x 128) = X(R x K) @ W(128 x K)^T + bias.  Wave = 16 rows x 128 cols.
// ---------------------------------------------------------------------------
template <int K>
__global__ void k_gemm_xw(const _Float16* __restrict__ X,
                          const _Float16* __restrict__ W16,
                          const float* __restrict__ bias,
                          _Float16* __restrict__ Y) {
  __shared__ _Float16 sW[HID * K];
  const int tid = threadIdx.x;
  {
    const uint4* src = (const uint4*)W16;
    uint4* dst = (uint4*)sW;
    for (int i = tid; i < HID * K / 8; i += 128) dst[i] = src[i];
  }
  __syncthreads();
  const int wave = tid >> 5, lane = tid & 31;
  const int hi = lane >> 4, lc = lane & 15;
  const long rowbase = (long)blockIdx.x * 64 + wave * 16;
  v8f acc[8];
#pragma unroll
  for (int nt = 0; nt < 8; ++nt) acc[nt] = splat8(bias[nt * 16 + lc]);
  const _Float16* xr = X + (rowbase + lc) * K;
#pragma unroll
  for (int kb = 0; kb < K; kb += 32) {
    v16h a = load_a_tile(xr + kb, hi);
#pragma unroll
    for (int nt = 0; nt < 8; ++nt) {
      v16h b = *(const v16h*)(sW + (long)(nt * 16 + lc) * K + kb + hi * 16);
      acc[nt] = wmma16(a, b, acc[nt]);
    }
  }
#pragma unroll
  for (int nt = 0; nt < 8; ++nt)
#pragma unroll
    for (int r = 0; r < 8; ++r)
      Y[(rowbase + r + hi * 8) * HID + nt * 16 + lc] = (_Float16)acc[nt][r];
}

// ---------------------------------------------------------------------------
// Z[bt] = relu( At(512x512) @ Y[bt](512x128) ).
// Block: 256 threads / 8 waves -> 128x128 output tile.  K staged in LDS
// TRANSPOSED (sYt[h][k], padded stride) + double-buffered: 1 barrier/K-step,
// B tiles = two aligned ds_load_b128 (no scalar u16 gathers).
// ---------------------------------------------------------------------------
__global__ void k_gemm_ax(const _Float16* __restrict__ At,
                          const _Float16* __restrict__ Yin,
                          _Float16* __restrict__ Z) {
  __shared__ _Float16 sYt[2][HID * AXPAD];
  const int tid = threadIdx.x;
  const int wave = tid >> 5, lane = tid & 31;
  const int hi = lane >> 4, lc = lane & 15;
  const int n0 = blockIdx.x * 128 + wave * 16;
  const long bt = blockIdx.y;
  const _Float16* Ybt = Yin + bt * NN * HID;
  const _Float16* ar = At + (long)(n0 + lc) * NN;

  // staging map: lane -> k (coalesced 32B per lane), wave -> 16-wide h slice;
  // LDS b16 stores hit consecutive banks across lanes.
  auto stage = [&](int kb, int buf) {
    const _Float16* src = Ybt + (long)(kb + lane) * HID + wave * 16;
    v8h a0 = *(const v8h*)src;
    v8h a1 = *(const v8h*)(src + 8);
#pragma unroll
    for (int j = 0; j < 8; ++j) {
      sYt[buf][(wave * 16 + j) * AXPAD + lane] = a0[j];
      sYt[buf][(wave * 16 + 8 + j) * AXPAD + lane] = a1[j];
    }
  };

  v8f acc[8];
#pragma unroll
  for (int c = 0; c < 8; ++c) acc[c] = splat8(0.f);

  stage(0, 0);
  __syncthreads();
  for (int it = 0; it < NN / 32; ++it) {
    const int kb = it * 32;
    if (kb + 32 < NN) {
      stage(kb + 32, (it + 1) & 1);
      if (kb + 64 < NN)
        __builtin_prefetch((const void*)(Ybt + (long)(kb + 64) * HID + tid * 16),
                           0, 0);
    }
    v16h a = load_a_tile(ar + kb, hi);
    const _Float16* sb = &sYt[it & 1][0];
#pragma unroll
    for (int c = 0; c < 8; ++c) {
      const _Float16* p = sb + (c * 16 + lc) * AXPAD + hi * 16;
      v8h blo = *(const v8h*)p;
      v8h bhi = *(const v8h*)(p + 8);
      acc[c] = wmma16(a, concat8(blo, bhi), acc[c]);
    }
    __syncthreads();
  }
#pragma unroll
  for (int c = 0; c < 8; ++c)
#pragma unroll
    for (int r = 0; r < 8; ++r) {
      float v = acc[c][r];
      v = v > 0.f ? v : 0.f;
      Z[(bt * NN + n0 + r + hi * 8) * HID + c * 16 + lc] = (_Float16)v;
    }
}

// ---------------------------------------------------------------------------
// GRU scan over T=64.  Wave owns 16 (b,n) rows; W_ih & W_hh (2x96KB f16) in LDS
// (CDNA5: 320KB/WGP).  h kept in regs (f32 C-layout + f16 A-layout), bias
// terms pre-hoisted into registers.
// ---------------------------------------------------------------------------
__global__ void k_gru_scan(const _Float16* __restrict__ Xin,
                           const _Float16* __restrict__ Wih16,
                           const _Float16* __restrict__ Whh16,
                           const float* __restrict__ b_ih,
                           const float* __restrict__ b_hh,
                           _Float16* __restrict__ H16,
                           float* __restrict__ H32) {
  extern __shared__ _Float16 smem[];
  _Float16* sWih = smem;                  // 384*128
  _Float16* sWhh = smem + 3 * HID * HID;  // 384*128
  _Float16* tscB = smem + 6 * HID * HID;  // 4 waves * 16*128 transpose scratch
  const int tid = threadIdx.x;
  {
    const uint4* s1 = (const uint4*)Wih16;
    const uint4* s2 = (const uint4*)Whh16;
    uint4* d1 = (uint4*)sWih;
    uint4* d2 = (uint4*)sWhh;
    for (int i = tid; i < 3 * HID * HID / 8; i += 128) { d1[i] = s1[i]; d2[i] = s2[i]; }
  }
  __syncthreads();
  const int wave = tid >> 5, lane = tid & 31;
  const int hi = lane >> 4, lc = lane & 15;
  _Float16* tsc = tscB + wave * 16 * HID;
  const int row = blockIdx.x * 64 + wave * 16;  // 0..4095 = b*512 + n
  const int bb = row >> 9;
  const int nn = row & 511;

  // hoist per-lane bias terms off the serial path
  float bR[8], bZ[8], bIN[8], bHN[8];
#pragma unroll
  for (int j = 0; j < 8; ++j) {
    const int c = j * 16 + lc;
    bR[j] = b_ih[c] + b_hh[c];
    bZ[j] = b_ih[HID + c] + b_hh[HID + c];
    bIN[j] = b_ih[2 * HID + c];
    bHN[j] = b_hh[2 * HID + c];
  }

  v8f h[8];
#pragma unroll
  for (int j = 0; j < 8; ++j) h[j] = splat8(0.f);
  v16h hA[4];
#pragma unroll
  for (int kb4 = 0; kb4 < 4; ++kb4)
#pragma unroll
    for (int j = 0; j < 16; ++j) hA[kb4][j] = (_Float16)0.f;

  for (int t = 0; t < NT; ++t) {
    const _Float16* xr = Xin + (((long)bb * NT + t) * NN + nn + lc) * HID;
    v16h xA[4];
#pragma unroll
    for (int kb4 = 0; kb4 < 4; ++kb4) xA[kb4] = load_a_tile(xr + kb4 * 32, hi);

#pragma unroll
    for (int j = 0; j < 8; ++j) {
      const int cR = j * 16 + lc;
      const int cZ = HID + cR;
      const int cN = 2 * HID + cR;
      v8f aR = splat8(bR[j]);
      v8f aZ = splat8(bZ[j]);
      v8f aIN = splat8(bIN[j]);
      v8f aHN = splat8(bHN[j]);
#pragma unroll
      for (int kb4 = 0; kb4 < 4; ++kb4) {
        const int ko = kb4 * 32 + hi * 16;
        v16h bRi = *(const v16h*)(sWih + (long)cR * HID + ko);
        v16h bRh = *(const v16h*)(sWhh + (long)cR * HID + ko);
        v16h bZi = *(const v16h*)(sWih + (long)cZ * HID + ko);
        v16h bZh = *(const v16h*)(sWhh + (long)cZ * HID + ko);
        v16h bNi = *(const v16h*)(sWih + (long)cN * HID + ko);
        v16h bNh = *(const v16h*)(sWhh + (long)cN * HID + ko);
        aR = wmma16(xA[kb4], bRi, aR);
        aR = wmma16(hA[kb4], bRh, aR);
        aZ = wmma16(xA[kb4], bZi, aZ);
        aZ = wmma16(hA[kb4], bZh, aZ);
        aIN = wmma16(xA[kb4], bNi, aIN);
        aHN = wmma16(hA[kb4], bNh, aHN);
      }
#pragma unroll
      for (int r = 0; r < 8; ++r) {
        float rg = sigf(aR[r]);
        float zg = sigf(aZ[r]);
        float ng = tanhf(aIN[r] + rg * aHN[r]);
        float hv = (1.0f - zg) * ng + zg * h[j][r];
        h[j][r] = hv;
        long idx = (((long)bb * NT + t) * NN + nn + r + hi * 8) * HID + j * 16 + lc;
        H16[idx] = (_Float16)hv;
        if (H32) H32[idx] = hv;
      }
    }
    // rebuild f16 A-layout copy of h via LDS transpose
    __syncthreads();
#pragma unroll
    for (int j = 0; j < 8; ++j)
#pragma unroll
      for (int r = 0; r < 8; ++r)
        tsc[(r + hi * 8) * HID + j * 16 + lc] = (_Float16)h[j][r];
    __syncthreads();
#pragma unroll
    for (int kb4 = 0; kb4 < 4; ++kb4)
      hA[kb4] = load_a_tile(tsc + (long)lc * HID + kb4 * 32, hi);
  }
}

// ---------------------------------------------------------------------------
// Readout: out = relu(h@W_r1^T + b1) @ W_r2^T + b2, fused per 16-row tile.
// ---------------------------------------------------------------------------
__global__ void k_readout(const _Float16* __restrict__ Hin,
                          const _Float16* __restrict__ Wr1,
                          const float* __restrict__ b_r1,
                          const _Float16* __restrict__ Wr2,
                          const float* __restrict__ b_r2,
                          float* __restrict__ Out) {
  extern __shared__ _Float16 smem[];
  _Float16* sW1 = smem;              // 128*128
  _Float16* sW2 = smem + HID * HID;  // 64*128
  _Float16* tscB = smem + HID * HID + NOUT * HID;
  const int tid = threadIdx.x;
  {
    const uint4* s1 = (const uint4*)Wr1;
    uint4* d1 = (uint4*)sW1;
    for (int i = tid; i < HID * HID / 8; i += 128) d1[i] = s1[i];
    const uint4* s2 = (const uint4*)Wr2;
    uint4* d2 = (uint4*)sW2;
    for (int i = tid; i < NOUT * HID / 8; i += 128) d2[i] = s2[i];
  }
  __syncthreads();
  const int wave = tid >> 5, lane = tid & 31;
  const int hi = lane >> 4, lc = lane & 15;
  _Float16* tsc = tscB + wave * 16 * HID;
  const long rowbase = (long)blockIdx.x * 64 + wave * 16;
  const _Float16* hr = Hin + (rowbase + lc) * HID;
  v16h hA[4];
#pragma unroll
  for (int kb4 = 0; kb4 < 4; ++kb4) hA[kb4] = load_a_tile(hr + kb4 * 32, hi);
#pragma unroll
  for (int j = 0; j < 8; ++j) {
    v8f acc = splat8(b_r1[j * 16 + lc]);
#pragma unroll
    for (int kb4 = 0; kb4 < 4; ++kb4) {
      v16h b = *(const v16h*)(sW1 + (long)(j * 16 + lc) * HID + kb4 * 32 + hi * 16);
      acc = wmma16(hA[kb4], b, acc);
    }
#pragma unroll
    for (int r = 0; r < 8; ++r) {
      float v = acc[r];
      v = v > 0.f ? v : 0.f;
      tsc[(r + hi * 8) * HID + j * 16 + lc] = (_Float16)v;
    }
  }
  __syncthreads();
  v16h rA[4];
#pragma unroll
  for (int kb4 = 0; kb4 < 4; ++kb4)
    rA[kb4] = load_a_tile(tsc + (long)lc * HID + kb4 * 32, hi);
#pragma unroll
  for (int jo = 0; jo < 4; ++jo) {
    v8f acc = splat8(b_r2[jo * 16 + lc]);
#pragma unroll
    for (int kb4 = 0; kb4 < 4; ++kb4) {
      v16h b = *(const v16h*)(sW2 + (long)(jo * 16 + lc) * HID + kb4 * 32 + hi * 16);
      acc = wmma16(rA[kb4], b, acc);
    }
#pragma unroll
    for (int r = 0; r < 8; ++r)
      Out[(rowbase + r + hi * 8) * NOUT + jo * 16 + lc] = acc[r];
  }
}

// ---------------------------------------------------------------------------
extern "C" void kernel_launch(void* const* d_in, const int* in_sizes, int n_in,
                              void* d_out, int out_size, void* d_ws, size_t ws_size,
                              hipStream_t stream) {
  const float* x_seq = (const float*)d_in[0];
  const float* A     = (const float*)d_in[1];
  const float* W_gc1 = (const float*)d_in[2];
  const float* b_gc1 = (const float*)d_in[3];
  const float* W_ih1 = (const float*)d_in[4];
  const float* W_hh1 = (const float*)d_in[5];
  const float* b_ih1 = (const float*)d_in[6];
  const float* b_hh1 = (const float*)d_in[7];
  const float* W_gc2 = (const float*)d_in[8];
  const float* b_gc2 = (const float*)d_in[9];
  const float* W_ih2 = (const float*)d_in[10];
  const float* W_hh2 = (const float*)d_in[11];
  const float* b_ih2 = (const float*)d_in[12];
  const float* b_hh2 = (const float*)d_in[13];
  const float* W_r1  = (const float*)d_in[14];
  const float* b_r1  = (const float*)d_in[15];
  const float* W_r2  = (const float*)d_in[16];
  const float* b_r2  = (const float*)d_in[17];
  (void)in_sizes; (void)n_in; (void)out_size; (void)ws_size;

  char* ws = (char*)d_ws;
  size_t off = 0;
  auto carve = [&](size_t bytes) -> char* {
    char* p = ws + off;
    off = (off + bytes + 255) & ~(size_t)255;
    return p;
  };
  float*    dis  = (float*)carve(NN * sizeof(float));
  _Float16* At16 = (_Float16*)carve((size_t)NN * NN * 2);
  _Float16* wgc1 = (_Float16*)carve(HID * F_IN * 2);
  _Float16* wih1 = (_Float16*)carve(3 * HID * HID * 2);
  _Float16* whh1 = (_Float16*)carve(3 * HID * HID * 2);
  _Float16* wgc2 = (_Float16*)carve(HID * HID * 2);
  _Float16* wih2 = (_Float16*)carve(3 * HID * HID * 2);
  _Float16* whh2 = (_Float16*)carve(3 * HID * HID * 2);
  _Float16* wr1  = (_Float16*)carve(HID * HID * 2);
  _Float16* wr2  = (_Float16*)carve(NOUT * HID * 2);
  _Float16* x16  = (_Float16*)carve((size_t)NROWS * F_IN * 2);
  _Float16* bufA = (_Float16*)carve((size_t)NROWS * HID * 2);
  _Float16* bufB = (_Float16*)carve((size_t)NROWS * HID * 2);

  auto cvt = [&](const float* s, _Float16* d, long n) {
    int grid = (int)((n + 255) / 256);
    if (grid > 4096) grid = 4096;
    k_f32_to_f16<<<grid, 256, 0, stream>>>(s, d, n);
  };
  cvt(W_gc1, wgc1, HID * F_IN);
  cvt(W_ih1, wih1, 3 * HID * HID);
  cvt(W_hh1, whh1, 3 * HID * HID);
  cvt(W_gc2, wgc2, HID * HID);
  cvt(W_ih2, wih2, 3 * HID * HID);
  cvt(W_hh2, whh2, 3 * HID * HID);
  cvt(W_r1, wr1, HID * HID);
  cvt(W_r2, wr2, NOUT * HID);
  cvt(x_seq, x16, (long)NROWS * F_IN);

  k_deg<<<NN, 256, 0, stream>>>(A, dis);
  k_At16<<<NN, 256, 0, stream>>>(A, dis, At16);

  float* out_y = (float*)d_out;                 // (B,T,N,OUT)
  float* out_h = out_y + (size_t)NROWS * NOUT;  // (B,T,N,HID)

  const size_t gruSmem = (size_t)(6 * HID * HID + 4 * 16 * HID) * 2;  // 212992 B
  const size_t roSmem  = (size_t)(HID * HID + NOUT * HID + 4 * 16 * HID) * 2;

  // block 1
  k_gemm_xw<F_IN><<<NROWS / 64, 128, 0, stream>>>(x16, wgc1, b_gc1, bufA);
  k_gemm_ax<<<dim3(NN / 128, NB * NT), 256, 0, stream>>>(At16, bufA, bufB);
  k_gru_scan<<<BNROWS / 64, 128, gruSmem, stream>>>(bufB, wih1, whh1, b_ih1, b_hh1,
                                                    bufA, nullptr);
  // block 2
  k_gemm_xw<HID><<<NROWS / 64, 128, 0, stream>>>(bufA, wgc2, b_gc2, bufB);
  k_gemm_ax<<<dim3(NN / 128, NB * NT), 256, 0, stream>>>(At16, bufB, bufA);
  k_gru_scan<<<BNROWS / 64, 128, gruSmem, stream>>>(bufA, wih2, whh2, b_ih2, b_hh2,
                                                    bufB, out_h);
  // readout
  k_readout<<<NROWS / 64, 128, roSmem, stream>>>(bufB, wr1, b_r1, wr2, b_r2, out_y);
}
